// Transformer_5334349382343
// MI455X (gfx1250) — matrix operand order repaired and checked by reading, not compiled, not run on hardware
//
#include <hip/hip_runtime.h>
#include <math.h>

// ---------------- model constants (match reference) ----------------
constexpr int Bsz = 8;
constexpr int Seq = 512;
constexpr int Dm  = 512;
constexpr int Hn  = 8;
constexpr int Dk  = 64;       // Dm / Hn
constexpr int Dff = 2048;
constexpr int Voc = 32000;
constexpr int NL  = 6;
constexpr int Mrows = Bsz * Seq;   // 4096
#define SQRT_D 22.62741699796952f  // sqrt(512)
#define NEG_INF_F (-__builtin_inff())

typedef __bf16 bf16_t;
typedef __bf16 v16bf __attribute__((ext_vector_type(16)));
typedef __bf16 v8bf  __attribute__((ext_vector_type(8)));
typedef float  v8f   __attribute__((ext_vector_type(8)));

// ---------------- WMMA fragment helpers (ISA §7.12.2 layouts) ----------------
// Both A (16x32) and B-from-transposed-source (32x16) fragments are two
// contiguous 16-byte chunks per lane, `gapElems` bf16 apart:
//   A frag:  gap = 8   (K 0..7 / 8..15 for lanes<16; 16..23 / 24..31 for >=16)
//   Bt frag: gap = 8   (K k0..k0+7 / k0+8..k0+15, k0 = kb + 16*(lane>=16))
__device__ inline v16bf load_pair(const bf16_t* __restrict__ p, int gapElems) {
  v8bf lo = *(const v8bf*)(p);
  v8bf hi = *(const v8bf*)(p + gapElems);
  v16bf f;
#pragma unroll
  for (int i = 0; i < 8; ++i) { f[i] = lo[i]; f[i + 8] = hi[i]; }
  return f;
}

__device__ inline v8f wmma_bf16(v16bf a, v16bf b, v8f c) {
  return __builtin_amdgcn_wmma_f32_16x16x32_bf16(false, a, false, b,
                                                 (short)0, c, false, false);
}

// issue the 8 WMMAs for one 32x64 k-step, consuming b0..b3 in load order
__device__ inline void mma_step(v8f acc[8], v16bf a0, v16bf a1,
                                v16bf b0, v16bf b1, v16bf b2, v16bf b3) {
  acc[0] = wmma_bf16(a0, b0, acc[0]);
  acc[4] = wmma_bf16(a1, b0, acc[4]);
  acc[1] = wmma_bf16(a0, b1, acc[1]);
  acc[5] = wmma_bf16(a1, b1, acc[5]);
  acc[2] = wmma_bf16(a0, b2, acc[2]);
  acc[6] = wmma_bf16(a1, b2, acc[6]);
  acc[3] = wmma_bf16(a0, b3, acc[3]);
  acc[7] = wmma_bf16(a1, b3, acc[7]);
}

// ---------------- generic GEMM: C[M,N] = A[M,K](bf16) @ W + bias --------------
// Weights pre-transposed: Wt[N,K] bf16 (K-contiguous rows). K is a template
// constant so all fragment offsets fold into instruction immediates; only two
// live address pairs (A base, B base) remain in the k-loop.
// grid = (M/32, N/256), block = 128 (4 waves); each wave computes 32x64.
// Software pipelined one k-step ahead.
template <int K>
__global__ __launch_bounds__(128, 1)
void gemm_kernel(const bf16_t* __restrict__ A,
                 const bf16_t* __restrict__ Wt,
                 const float* __restrict__ bias,
                 float* __restrict__ C, bf16_t* __restrict__ Cb,
                 int M, int N, int relu) {
  int mt   = blockIdx.x * 32;
  int wave = threadIdx.x >> 5;
  int nt   = blockIdx.y * 256 + wave * 64;
  int lane = threadIdx.x & 31;
  int l15 = lane & 15, hh = lane >> 4;

  const bf16_t* aB = A  + (size_t)(mt + l15) * K + hh * 16;  // row mt+l15
  const bf16_t* bB = Wt + (size_t)(nt + l15) * K + hh * 16;  // col nt+l15

  v8f acc[8] = {};   // [mi*4 + j]
  v16bf a0 = load_pair(aB,                 8);
  v16bf a1 = load_pair(aB + 16 * K,        8);
  v16bf b0 = load_pair(bB,                 8);
  v16bf b1 = load_pair(bB + 16 * K,        8);
  v16bf b2 = load_pair(bB + 32 * K,        8);
  v16bf b3 = load_pair(bB + 48 * K,        8);
  for (int k = 32; k < K; k += 32) {
    v16bf na0 = load_pair(aB + k,               8);
    v16bf na1 = load_pair(aB + 16 * K + k,      8);
    v16bf nb0 = load_pair(bB + k,               8);
    v16bf nb1 = load_pair(bB + 16 * K + k,      8);
    v16bf nb2 = load_pair(bB + 32 * K + k,      8);
    v16bf nb3 = load_pair(bB + 48 * K + k,      8);
    mma_step(acc, a0, a1, b0, b1, b2, b3);
    a0 = na0; a1 = na1; b0 = nb0; b1 = nb1; b2 = nb2; b3 = nb3;
  }
  mma_step(acc, a0, a1, b0, b1, b2, b3);

#pragma unroll
  for (int mi = 0; mi < 2; ++mi) {
#pragma unroll
    for (int j = 0; j < 4; ++j) {
      int col = nt + j * 16 + l15;
      float bv = bias ? bias[col] : 0.f;
#pragma unroll
      for (int i = 0; i < 8; ++i) {
        float v = acc[mi * 4 + j][i] + bv;
        if (relu) v = fmaxf(v, 0.f);
        size_t idx = (size_t)(mt + mi * 16 + hh * 8 + i) * N + col;
        if (C)  C[idx]  = v;
        if (Cb) Cb[idx] = (bf16_t)v;
      }
    }
  }
}

// ---------------- QK^T scores with masks --------------------------------------
// grid = (B*H, S/32, S/256), block = 128. Q,K: [B*S, D] bf16, head offset h*Dk.
__global__ __launch_bounds__(128, 1)
void qk_kernel(const bf16_t* __restrict__ Q,
               const bf16_t* __restrict__ Km,
               const float* __restrict__ pad, int causal,
               float* __restrict__ Sc) {
  int bh = blockIdx.x;
  int b = bh / Hn, h = bh % Hn;
  int q0   = blockIdx.y * 32;
  int wave = threadIdx.x >> 5;
  int k0   = blockIdx.z * 256 + wave * 64;
  int lane = threadIdx.x & 31;
  int l15 = lane & 15, hh = lane >> 4;

  const bf16_t* aB = Q  + ((size_t)b * Seq + q0 + l15) * Dm + h * Dk + hh * 16;
  const bf16_t* bB = Km + ((size_t)b * Seq + k0 + l15) * Dm + h * Dk + hh * 16;

  v8f acc[8] = {};
  v16bf a0 = load_pair(aB,                 8);
  v16bf a1 = load_pair(aB + 16 * Dm,       8);
  v16bf b0 = load_pair(bB,                 8);
  v16bf b1 = load_pair(bB + 16 * Dm,       8);
  v16bf b2 = load_pair(bB + 32 * Dm,       8);
  v16bf b3 = load_pair(bB + 48 * Dm,       8);
  for (int kk = 32; kk < Dk; kk += 32) {
    v16bf na0 = load_pair(aB + kk,              8);
    v16bf na1 = load_pair(aB + 16 * Dm + kk,    8);
    v16bf nb0 = load_pair(bB + kk,              8);
    v16bf nb1 = load_pair(bB + 16 * Dm + kk,    8);
    v16bf nb2 = load_pair(bB + 32 * Dm + kk,    8);
    v16bf nb3 = load_pair(bB + 48 * Dm + kk,    8);
    mma_step(acc, a0, a1, b0, b1, b2, b3);
    a0 = na0; a1 = na1; b0 = nb0; b1 = nb1; b2 = nb2; b3 = nb3;
  }
  mma_step(acc, a0, a1, b0, b1, b2, b3);

#pragma unroll
  for (int mi = 0; mi < 2; ++mi) {
#pragma unroll
    for (int j = 0; j < 4; ++j) {
      int key = k0 + j * 16 + l15;
      float pv = pad[b * Seq + key];
#pragma unroll
      for (int i = 0; i < 8; ++i) {
        int qr = q0 + mi * 16 + hh * 8 + i;
        float v = acc[mi * 4 + j][i] * 0.125f + pv;   // 1/sqrt(64)
        if (causal && key > qr) v = NEG_INF_F;
        Sc[((size_t)bh * Seq + qr) * Seq + key] = v;
      }
    }
  }
}

// ---------------- row softmax -> bf16 probs -----------------------------------
// grid = B*H*S, block = 256 (2 elements/thread over S=512)
__global__ void softmax_kernel(const float* __restrict__ Sc,
                               bf16_t* __restrict__ P) {
  __shared__ float red[256];
  size_t row = blockIdx.x;
  const float* s = Sc + row * Seq;
  int t = threadIdx.x;
  float v0 = s[t], v1 = s[t + 256];
  red[t] = fmaxf(v0, v1);
  __syncthreads();
  for (int off = 128; off > 0; off >>= 1) {
    if (t < off) red[t] = fmaxf(red[t], red[t + off]);
    __syncthreads();
  }
  float m = red[0];
  __syncthreads();
  float e0, e1;
  if (m < -1e37f) { e0 = 0.f; e1 = 0.f; }
  else { e0 = __expf(v0 - m); e1 = __expf(v1 - m); }
  red[t] = e0 + e1;
  __syncthreads();
  for (int off = 128; off > 0; off >>= 1) {
    if (t < off) red[t] += red[t + off];
    __syncthreads();
  }
  float inv = red[0] > 0.f ? 1.f / red[0] : 0.f;
  P[row * Seq + t]       = (bf16_t)(e0 * inv);
  P[row * Seq + t + 256] = (bf16_t)(e1 * inv);
}

// ---------------- O = P @ V using transposed V ([B, D, S]) --------------------
// grid = (B*H, S/128), block = 128; wave w handles a 32-query tile x 64 dims.
__global__ __launch_bounds__(128, 1)
void av_kernel(const bf16_t* __restrict__ P,
               const bf16_t* __restrict__ Vt,
               bf16_t* __restrict__ O) {
  int bh = blockIdx.x;
  int b = bh / Hn, h = bh % Hn;
  int wave = threadIdx.x >> 5;
  int q0 = blockIdx.y * 128 + wave * 32;
  int lane = threadIdx.x & 31;
  int l15 = lane & 15, hh = lane >> 4;

  const bf16_t* aB = P  + ((size_t)bh * Seq + q0 + l15) * Seq + hh * 16;
  const bf16_t* bB = Vt + ((size_t)b * Dm + h * Dk + l15) * Seq + hh * 16;

  v8f acc[8] = {};
  v16bf a0 = load_pair(aB,                 8);
  v16bf a1 = load_pair(aB + 16 * Seq,      8);
  v16bf b0 = load_pair(bB,                 8);
  v16bf b1 = load_pair(bB + 16 * Seq,      8);
  v16bf b2 = load_pair(bB + 32 * Seq,      8);
  v16bf b3 = load_pair(bB + 48 * Seq,      8);
  for (int k = 32; k < Seq; k += 32) {
    v16bf na0 = load_pair(aB + k,               8);
    v16bf na1 = load_pair(aB + 16 * Seq + k,    8);
    v16bf nb0 = load_pair(bB + k,               8);
    v16bf nb1 = load_pair(bB + 16 * Seq + k,    8);
    v16bf nb2 = load_pair(bB + 32 * Seq + k,    8);
    v16bf nb3 = load_pair(bB + 48 * Seq + k,    8);
    mma_step(acc, a0, a1, b0, b1, b2, b3);
    a0 = na0; a1 = na1; b0 = nb0; b1 = nb1; b2 = nb2; b3 = nb3;
  }
  mma_step(acc, a0, a1, b0, b1, b2, b3);

#pragma unroll
  for (int mi = 0; mi < 2; ++mi) {
#pragma unroll
    for (int j = 0; j < 4; ++j) {
      int col = h * Dk + j * 16 + l15;
#pragma unroll
      for (int i = 0; i < 8; ++i) {
        int row = b * Seq + q0 + mi * 16 + hh * 8 + i;
        O[(size_t)row * Dm + col] = (bf16_t)acc[mi * 4 + j][i];
      }
    }
  }
}

// ---------------- fp32 [K,N] -> bf16 transposed [N,K] (LDS tiled) -------------
// grid = (N/32, K/32), block = (32, 8)
__global__ void convertT_kernel(const float* __restrict__ src,
                                bf16_t* __restrict__ dst, int K, int N) {
  __shared__ float tile[32][33];
  int n0 = blockIdx.x * 32, k0 = blockIdx.y * 32;
  for (int r = threadIdx.y; r < 32; r += 8)
    tile[r][threadIdx.x] = src[(size_t)(k0 + r) * N + n0 + threadIdx.x];
  __syncthreads();
  for (int r = threadIdx.y; r < 32; r += 8)
    dst[(size_t)(n0 + r) * K + k0 + threadIdx.x] = (bf16_t)tile[threadIdx.x][r];
}

// ---------------- bf16 [B,S,D] -> [B,D,S] transpose (for V) -------------------
// grid = (S/32, D/32, B), block = (32, 8)
__global__ void transpose_bf16_kernel(const bf16_t* __restrict__ src,
                                      bf16_t* __restrict__ dst) {
  __shared__ bf16_t tile[32][33];
  int b = blockIdx.z;
  int s0 = blockIdx.x * 32, d0 = blockIdx.y * 32;
  const bf16_t* S = src + (size_t)b * Seq * Dm;
  bf16_t*       D = dst + (size_t)b * Dm * Seq;
  for (int r = threadIdx.y; r < 32; r += 8)
    tile[r][threadIdx.x] = S[(size_t)(s0 + r) * Dm + d0 + threadIdx.x];
  __syncthreads();
  for (int r = threadIdx.y; r < 32; r += 8)
    D[(size_t)(d0 + r) * Seq + s0 + threadIdx.x] = tile[threadIdx.x][r];
}

// ---------------- embedding + positional encoding + pad mask -----------------
// grid = B*S, block = 256
__global__ void embed_kernel(const int* __restrict__ tok,
                             const float* __restrict__ emb,
                             float* __restrict__ X, bf16_t* __restrict__ Xb,
                             float* __restrict__ pad) {
  int bs = blockIdx.x;
  int s  = bs % Seq;
  int t  = tok[bs];
  if (threadIdx.x == 0) pad[bs] = (t == 0) ? NEG_INF_F : 0.f;
  for (int d = threadIdx.x; d < Dm; d += blockDim.x) {
    float ang = (float)s * __powf(10000.f, -2.f * (float)d / (float)Dm);
    float pe  = (d & 1) ? __cosf(ang) : __sinf(ang);
    float v   = emb[(size_t)t * Dm + d] * SQRT_D + pe;
    X[(size_t)bs * Dm + d]  = v;
    Xb[(size_t)bs * Dm + d] = (bf16_t)v;
  }
}

// ---------------- residual add + LayerNorm (in-place safe) -------------------
// grid = B*S rows, block = 256 (2 elems/thread over D=512)
__global__ void add_ln_kernel(const float* __restrict__ X,
                              const float* __restrict__ Sub,
                              const float* __restrict__ g,
                              const float* __restrict__ bta,
                              float* __restrict__ Y, bf16_t* __restrict__ Yb) {
  __shared__ float red[256];
  int row = blockIdx.x, t = threadIdx.x;
  const float* x = X   + (size_t)row * Dm;
  const float* s = Sub + (size_t)row * Dm;
  float v0 = x[t] + s[t];
  float v1 = x[t + 256] + s[t + 256];
  red[t] = v0 + v1;
  __syncthreads();
  for (int off = 128; off > 0; off >>= 1) {
    if (t < off) red[t] += red[t + off];
    __syncthreads();
  }
  float mu = red[0] * (1.f / Dm);
  __syncthreads();
  float d0 = v0 - mu, d1 = v1 - mu;
  red[t] = d0 * d0 + d1 * d1;
  __syncthreads();
  for (int off = 128; off > 0; off >>= 1) {
    if (t < off) red[t] += red[t + off];
    __syncthreads();
  }
  float rstd = rsqrtf(red[0] * (1.f / Dm) + 1e-5f);
  float y0 = d0 * rstd * g[t] + bta[t];
  float y1 = d1 * rstd * g[t + 256] + bta[t + 256];
  Y[(size_t)row * Dm + t]        = y0;
  Y[(size_t)row * Dm + t + 256]  = y1;
  Yb[(size_t)row * Dm + t]       = (bf16_t)y0;
  Yb[(size_t)row * Dm + t + 256] = (bf16_t)y1;
}

// ================================ host side ===================================

struct MhaW { const float *bq, *bk, *bv, *bo; bf16_t *wq, *wk, *wv, *wo; };
struct FfnW { const float *b1, *b2; bf16_t *w1, *w2; };
struct LnW  { const float *g, *b; };

extern "C" void kernel_launch(void* const* d_in, const int* in_sizes, int n_in,
                              void* d_out, int out_size, void* d_ws, size_t ws_size,
                              hipStream_t stream) {
  (void)in_sizes; (void)n_in; (void)out_size; (void)ws_size;

  // ---- workspace bump allocator
  char* wsp = (char*)d_ws;
  size_t off = 0;
  auto alloc = [&](size_t bytes) -> void* {
    void* p = wsp + off;
    off = (off + bytes + 255) & ~(size_t)255;
    return p;
  };
  // convert + transpose weight: src fp32 [K,N] -> dst bf16 [N,K]
  auto tobfT = [&](const float* src, int K, int N) -> bf16_t* {
    bf16_t* dst = (bf16_t*)alloc((size_t)K * N * sizeof(bf16_t));
    convertT_kernel<<<dim3(N / 32, K / 32), dim3(32, 8), 0, stream>>>(src, dst, K, N);
    return dst;
  };

  // ---- parse inputs (setup_inputs dict order, nested depth-first)
  int idx = 0;
  const int* src_tok = (const int*)d_in[idx++];
  const int* tgt_tok = (const int*)d_in[idx++];
  const float* emb_in  = (const float*)d_in[idx++];
  const float* emb_out = (const float*)d_in[idx++];

  auto parse_mha = [&](MhaW& m) {
    const float* wq = (const float*)d_in[idx++]; m.bq = (const float*)d_in[idx++];
    const float* wk = (const float*)d_in[idx++]; m.bk = (const float*)d_in[idx++];
    const float* wv = (const float*)d_in[idx++]; m.bv = (const float*)d_in[idx++];
    const float* wo = (const float*)d_in[idx++]; m.bo = (const float*)d_in[idx++];
    m.wq = tobfT(wq, Dm, Dm); m.wk = tobfT(wk, Dm, Dm);
    m.wv = tobfT(wv, Dm, Dm); m.wo = tobfT(wo, Dm, Dm);
  };
  auto parse_ffn = [&](FfnW& f) {
    const float* w1 = (const float*)d_in[idx++]; f.b1 = (const float*)d_in[idx++];
    const float* w2 = (const float*)d_in[idx++]; f.b2 = (const float*)d_in[idx++];
    f.w1 = tobfT(w1, Dm, Dff);  // -> [Dff, Dm]
    f.w2 = tobfT(w2, Dff, Dm);  // -> [Dm, Dff]
  };
  auto parse_ln = [&](LnW& l) {
    l.g = (const float*)d_in[idx++]; l.b = (const float*)d_in[idx++];
  };

  MhaW enc_attn[NL]; FfnW enc_ffn[NL]; LnW enc_ln[NL][2];
  for (int l = 0; l < NL; ++l) {
    parse_mha(enc_attn[l]); parse_ffn(enc_ffn[l]);
    parse_ln(enc_ln[l][0]); parse_ln(enc_ln[l][1]);
  }
  MhaW dec_sa[NL], dec_ca[NL]; FfnW dec_ffn[NL]; LnW dec_ln[NL][3];
  for (int l = 0; l < NL; ++l) {
    parse_mha(dec_sa[l]); parse_mha(dec_ca[l]); parse_ffn(dec_ffn[l]);
    parse_ln(dec_ln[l][0]); parse_ln(dec_ln[l][1]); parse_ln(dec_ln[l][2]);
  }
  const float* proj_w = (const float*)d_in[idx++];
  const float* proj_b = (const float*)d_in[idx++];
  bf16_t* proj_wt = tobfT(proj_w, Dm, Voc);   // -> [Voc, Dm]

  // ---- activation buffers
  float*  padS = (float*)alloc(Mrows * sizeof(float));
  float*  padT = (float*)alloc(Mrows * sizeof(float));
  float*  Xf = (float*) alloc((size_t)Mrows * Dm * sizeof(float));
  bf16_t* Xb = (bf16_t*)alloc((size_t)Mrows * Dm * sizeof(bf16_t));
  float*  Yf = (float*) alloc((size_t)Mrows * Dm * sizeof(float));
  bf16_t* Yb = (bf16_t*)alloc((size_t)Mrows * Dm * sizeof(bf16_t));
  bf16_t* Qb = (bf16_t*)alloc((size_t)Mrows * Dm * sizeof(bf16_t));
  bf16_t* Kb = (bf16_t*)alloc((size_t)Mrows * Dm * sizeof(bf16_t));
  bf16_t* Vb = (bf16_t*)alloc((size_t)Mrows * Dm * sizeof(bf16_t));
  bf16_t* Vt = (bf16_t*)alloc((size_t)Mrows * Dm * sizeof(bf16_t));  // [B, D, S]
  bf16_t* Ob = (bf16_t*)alloc((size_t)Mrows * Dm * sizeof(bf16_t));
  float*  Sub = (float*)alloc((size_t)Mrows * Dm * sizeof(float));
  bf16_t* Hb = (bf16_t*)alloc((size_t)Mrows * Dff * sizeof(bf16_t));
  float*  Sc = (float*) alloc((size_t)Bsz * Hn * Seq * Seq * sizeof(float));
  bf16_t* Pb = (bf16_t*)alloc((size_t)Bsz * Hn * Seq * Seq * sizeof(bf16_t));

  auto gemm = [&](const bf16_t* A, const bf16_t* Wt, const float* bias,
                  float* C, bf16_t* Cb, int N, int K, int relu) {
    dim3 grid(Mrows / 32, N / 256);
    if (K == 512)
      gemm_kernel<512><<<grid, 128, 0, stream>>>(A, Wt, bias, C, Cb, Mrows, N, relu);
    else
      gemm_kernel<2048><<<grid, 128, 0, stream>>>(A, Wt, bias, C, Cb, Mrows, N, relu);
  };

  auto run_mha = [&](const MhaW& m, const bf16_t* xq, const bf16_t* xkv,
                     const float* pad, int causal, float* subOut) {
    gemm(xq,  m.wq, m.bq, nullptr, Qb, Dm, Dm, 0);
    gemm(xkv, m.wk, m.bk, nullptr, Kb, Dm, Dm, 0);
    gemm(xkv, m.wv, m.bv, nullptr, Vb, Dm, Dm, 0);
    transpose_bf16_kernel<<<dim3(Seq / 32, Dm / 32, Bsz), dim3(32, 8), 0, stream>>>(Vb, Vt);
    qk_kernel<<<dim3(Bsz * Hn, Seq / 32, Seq / 256), 128, 0, stream>>>(
        Qb, Kb, pad, causal, Sc);
    softmax_kernel<<<Bsz * Hn * Seq, 256, 0, stream>>>(Sc, Pb);
    av_kernel<<<dim3(Bsz * Hn, Seq / 128), 128, 0, stream>>>(Pb, Vt, Ob);
    gemm(Ob, m.wo, m.bo, subOut, nullptr, Dm, Dm, 0);
  };
  auto run_ffn = [&](const FfnW& f, const bf16_t* x, float* subOut) {
    gemm(x,  f.w1, f.b1, nullptr, Hb, Dff, Dm, 1);   // relu
    gemm(Hb, f.w2, f.b2, subOut, nullptr, Dm, Dff, 0);
  };
  auto add_ln = [&](float* X, bf16_t* XB, const float* sub, const LnW& ln) {
    add_ln_kernel<<<Mrows, 256, 0, stream>>>(X, sub, ln.g, ln.b, X, XB);
  };

  // ---- embeddings (+ pad masks)
  embed_kernel<<<Mrows, 256, 0, stream>>>(src_tok, emb_in,  Xf, Xb, padS);
  embed_kernel<<<Mrows, 256, 0, stream>>>(tgt_tok, emb_out, Yf, Yb, padT);

  // ---- encoder
  for (int l = 0; l < NL; ++l) {
    run_mha(enc_attn[l], Xb, Xb, padS, /*causal=*/0, Sub);
    add_ln(Xf, Xb, Sub, enc_ln[l][0]);
    run_ffn(enc_ffn[l], Xb, Sub);
    add_ln(Xf, Xb, Sub, enc_ln[l][1]);
  }
  // Xb now holds enc_out (bf16); decoder never overwrites it.

  // ---- decoder
  for (int l = 0; l < NL; ++l) {
    run_mha(dec_sa[l], Yb, Yb, padT, /*causal=*/1, Sub);
    add_ln(Yf, Yb, Sub, dec_ln[l][0]);
    run_mha(dec_ca[l], Yb, Xb, padS, /*causal=*/0, Sub);
    add_ln(Yf, Yb, Sub, dec_ln[l][1]);
    run_ffn(dec_ffn[l], Yb, Sub);
    add_ln(Yf, Yb, Sub, dec_ln[l][2]);
  }

  // ---- output projection: logits (B,S,V) fp32 directly into d_out
  gemm(Yb, proj_wt, proj_b, (float*)d_out, nullptr, Voc, Dm, 0);
}
